// DeconvLayer_8890582303190
// MI455X (gfx1250) — compile-verified, matrix operand order checked by reading, not run on hardware
//
#include <hip/hip_runtime.h>
#include <hip/hip_bf16.h>
#include <math.h>

// ---------------------------------------------------------------------------
// DeformConv + BN/ReLU + ConvTranspose2d(4x4,s2) + BN/ReLU for MI455X gfx1250
// GEMMs: v_wmma_f32_16x16x32_f16 (fp32 accumulate), B staged in LDS (double
// buffered, shared by the 8 waves of a workgroup).
// ---------------------------------------------------------------------------

typedef __attribute__((ext_vector_type(16))) _Float16 v16h;
typedef __attribute__((ext_vector_type(8)))  float    v8f;

#define B_    16
#define CIN_  256
#define COUT_ 128
#define H_    32
#define W_    32
#define HW_   (H_ * W_)
#define P_    (B_ * HW_)        // 16384 pixels
#define KPIX_ 9                 // 3x3 taps
#define KDCN_ (CIN_ * KPIX_)    // 2304
#define NOFF_ 27                // offset-conv real channels
#define NOFFP 32                // padded to 16-multiple
#define KUP_  512               // CIN(128) * 4 taps per parity class
#define POUT_ 65536             // 16 * 64 * 64 output pixels
#define EPS_  1e-5f

// ---- workspace layout (bytes), all offsets multiples of 256 ----
#define OFF_BUFA   0ull                              // 16384*2304*2 = 75497472 (reused 3x)
#define OFF_OFFC   75497472ull                       // 16384*32*4   = 2097152
#define OFF_WOFF   77594624ull                       // 32*2304*2    = 147456
#define OFF_WDCN   77742080ull                       // 128*2304*2   = 589824
#define OFF_WUP    78331904ull                       // 4*128*512*2  = 524288
#define OFF_YPRE   78856192ull                       // 16384*128*4  = 8388608
#define OFF_YF16   87244800ull                       // 16*128*1024*2= 4194304
#define OFF_OUTPRE 91439104ull                       // 65536*128*4  = 33554432
#define OFF_STATS  124993536ull                      // 512*4        = 2048

__device__ __forceinline__ int iclamp(int v, int lo, int hi) {
    return v < lo ? lo : (v > hi ? hi : v);
}

// ===========================================================================
// WMMA GEMM with LDS-staged B:
//   C[M][NT*16] = A[M][K] (f16 row-major) * B (f16 stored as Bt[N][K]) + bias
// Block = 256 threads = 8 waves = 8 consecutive 16-row strips sharing one B.
// B is double-buffered in LDS in 32-K chunks (NT*16 x 32 halves per chunk).
// ===========================================================================
template <int NT>
__global__ void gemm_wmma_f16(const _Float16* __restrict__ A,
                              const _Float16* __restrict__ Bt,
                              float* __restrict__ C,
                              const float* __restrict__ bias, int biasN,
                              int M, int K) {
    constexpr int L = NT * 16;                 // B rows (output channels)
    __shared__ _Float16 Bs[2][L][32];          // 2 * L * 64 bytes

    const int tid    = (int)threadIdx.x;
    const int wave   = (int)((blockIdx.x * blockDim.x + tid) >> 5);
    const int lane   = tid & 31;
    const int half16 = lane >> 4;              // 0: K 0..15, 1: K 16..31
    const int lrow   = lane & 15;
    const int row0   = wave << 4;

    // cooperative B stager: NT*32 v16h vectors per chunk, one per thread
    const int sn = tid % L;                    // B row
    const int sh = tid / L;                    // K half (0/1) when tid < 2*L
    const bool stager = (tid < 2 * L);

    v8f acc[NT];
    const v8f zero = {0.f, 0.f, 0.f, 0.f, 0.f, 0.f, 0.f, 0.f};
#pragma unroll
    for (int nt = 0; nt < NT; ++nt) acc[nt] = zero;

    const _Float16* aRow = A + (size_t)(row0 + lrow) * K + (half16 << 4);

    // prime buffer 0
    if (stager)
        *(v16h*)&Bs[0][sn][sh << 4] =
            *(const v16h*)(Bt + (size_t)sn * K + (sh << 4));
    __syncthreads();

    int buf = 0;
#pragma unroll 1
    for (int k0 = 0; k0 < K; k0 += 32) {
        // stage next chunk into the other buffer
        if (k0 + 32 < K && stager)
            *(v16h*)&Bs[buf ^ 1][sn][sh << 4] =
                *(const v16h*)(Bt + (size_t)sn * K + k0 + 32 + (sh << 4));

        v16h av = *(const v16h*)(aRow + k0);
#pragma unroll
        for (int nt = 0; nt < NT; ++nt) {
            v16h bv = *(const v16h*)&Bs[buf][(nt << 4) + lrow][half16 << 4];
            acc[nt] = __builtin_amdgcn_wmma_f32_16x16x32_f16(
                false, av, false, bv, (short)0, acc[nt], false, false);
        }
        __syncthreads();
        buf ^= 1;
    }

    // C/D layout: element e of v8f -> row = row0 + e + half16*8, col = lrow+nt*16
    const int crow = row0 + (half16 << 3);
#pragma unroll
    for (int nt = 0; nt < NT; ++nt) {
        const int col = (nt << 4) + lrow;
        float badd = (bias != nullptr && col < biasN) ? bias[col] : 0.0f;
#pragma unroll
        for (int e = 0; e < 8; ++e) {
            C[(size_t)(crow + e) * (NT * 16) + col] = acc[nt][e] + badd;
        }
    }
}

// ===========================================================================
// Weight prep: fp32 -> f16, packed as Bt[N][K]
// ===========================================================================
__global__ void prep_woff(const float* __restrict__ w, _Float16* __restrict__ o) {
    int idx = blockIdx.x * blockDim.x + threadIdx.x;     // NOFFP * KDCN_
    if (idx >= NOFFP * KDCN_) return;
    int n = idx / KDCN_, k = idx - n * KDCN_;
    float v = (n < NOFF_) ? w[(size_t)n * KDCN_ + k] : 0.0f;
    o[idx] = (_Float16)v;                                // K-order: c*9+k
}

// DCN weights re-packed with K-order k*256+c (matches dcn_sample's layout)
__global__ void prep_wdcn(const float* __restrict__ w, _Float16* __restrict__ o) {
    int idx = blockIdx.x * blockDim.x + threadIdx.x;     // COUT_ * KDCN_
    if (idx >= COUT_ * KDCN_) return;
    int c = idx & (CIN_ - 1);
    int k = (idx >> 8) % KPIX_;
    int n = idx / KDCN_;
    o[idx] = (_Float16)w[((size_t)n * CIN_ + c) * KPIX_ + k];
}

// transposed-conv weights: 4 parity classes, Bt[pc][co][ci*4 + jj*2 + ll]
//   j = (1-py) + 2*jj ; l = (1-px) + 2*ll ; value = w_up[ci][co][j][l]
__global__ void prep_wup(const float* __restrict__ w, _Float16* __restrict__ o) {
    int idx = blockIdx.x * blockDim.x + threadIdx.x;     // 4*128*512
    if (idx >= 4 * COUT_ * KUP_) return;
    int kk = idx & (KUP_ - 1);
    int co = (idx >> 9) & (COUT_ - 1);
    int pc = idx >> 16;
    int ci = kk >> 2, jj = (kk >> 1) & 1, ll = kk & 1;
    int py = pc >> 1, px = pc & 1;
    int j = (1 - py) + 2 * jj;
    int l = (1 - px) + 2 * ll;
    o[idx] = (_Float16)w[(((size_t)ci * COUT_ + co) * 4 + j) * 4 + l];
}

__global__ void zero_stats(float* __restrict__ s) {
    int idx = blockIdx.x * blockDim.x + threadIdx.x;
    if (idx < 512) s[idx] = 0.0f;
}

// ===========================================================================
// im2col of x into f16 patches: patch[p][c*9 + ky*3+kx] (zero padded)
// ===========================================================================
__global__ void im2col_x(const float* __restrict__ x, _Float16* __restrict__ patch) {
    long long idx = (long long)blockIdx.x * blockDim.x + threadIdx.x;
    if (idx >= (long long)P_ * KDCN_) return;
    int t = (int)(idx % KDCN_);
    int p = (int)(idx / KDCN_);
    int c = t / KPIX_, k = t - c * KPIX_;
    int b = p >> 10, rem = p & 1023, yy = rem >> 5, xx = rem & 31;
    int iy = yy + k / 3 - 1, ix = xx + k % 3 - 1;
    float v = 0.0f;
    if (iy >= 0 && iy < H_ && ix >= 0 && ix < W_)
        v = x[(((size_t)b * CIN_ + c) * H_ + iy) * W_ + ix];
    patch[idx] = (_Float16)v;
}

// ===========================================================================
// Deformable bilinear sampling: one thread per (pixel, tap); loops channels.
// offC layout: [p][32] fp32 (channels 0..26 valid).
//   dy = ch 2k, dx = ch 2k+1, mask = sigmoid(ch 18+k)
// samp[p][k*256+c] = bilinear(x, p, tap k, chan c) * mask   (f16, contiguous
// 512B burst per thread; K-order matches prep_wdcn)
// ===========================================================================
__global__ void dcn_sample(const float* __restrict__ x,
                           const float* __restrict__ offC,
                           _Float16* __restrict__ samp) {
    int idx = blockIdx.x * blockDim.x + threadIdx.x;     // P_ * 9
    if (idx >= P_ * KPIX_) return;
    int p = idx / KPIX_, k = idx - p * KPIX_;
    int b = p >> 10, rem = p & 1023, yy = rem >> 5, xx = rem & 31;

    const float* o = offC + (size_t)p * NOFFP;
    float dy = o[2 * k], dx = o[2 * k + 1];
    float m  = 1.0f / (1.0f + expf(-o[18 + k]));

    float py = (float)(yy + k / 3 - 1) + dy;
    float px = (float)(xx + k % 3 - 1) + dx;
    float y0f = floorf(py), x0f = floorf(px);
    float wy = py - y0f, wx = px - x0f;
    int y0 = (int)y0f, x0 = (int)x0f;
    int y1 = y0 + 1, x1 = x0 + 1;

    bool vy0 = (y0 >= 0 && y0 <= H_ - 1), vy1 = (y1 >= 0 && y1 <= H_ - 1);
    bool vx0 = (x0 >= 0 && x0 <= W_ - 1), vx1 = (x1 >= 0 && x1 <= W_ - 1);
    float w00 = (vy0 && vx0) ? (1.f - wy) * (1.f - wx) : 0.f;
    float w01 = (vy0 && vx1) ? (1.f - wy) * wx         : 0.f;
    float w10 = (vy1 && vx0) ? wy * (1.f - wx)         : 0.f;
    float w11 = (vy1 && vx1) ? wy * wx                 : 0.f;

    int y0c = iclamp(y0, 0, H_ - 1), y1c = iclamp(y1, 0, H_ - 1);
    int x0c = iclamp(x0, 0, W_ - 1), x1c = iclamp(x1, 0, W_ - 1);
    int i00 = y0c * W_ + x0c, i01 = y0c * W_ + x1c;
    int i10 = y1c * W_ + x0c, i11 = y1c * W_ + x1c;

    const float* xb = x + (size_t)b * CIN_ * HW_;
    _Float16* sp = samp + (size_t)p * KDCN_ + (size_t)k * CIN_;
#pragma unroll 4
    for (int c = 0; c < CIN_; ++c) {
        const float* xc = xb + (size_t)c * HW_;
        float v = xc[i00] * w00 + xc[i01] * w01 + xc[i10] * w10 + xc[i11] * w11;
        sp[c] = (_Float16)(v * m);
    }
}

// ===========================================================================
// BN statistics: pre[Mrows][128] -> stats[0..127]=sum, [128..255]=sumsq
// block = 128 threads (one channel each), 64 rows per block, fp32 atomics.
// ===========================================================================
__global__ void bn_stats(const float* __restrict__ pre, int Mrows,
                         float* __restrict__ stats) {
    int c  = (int)threadIdx.x;
    int r0 = (int)blockIdx.x * 64;
    int r1 = r0 + 64; if (r1 > Mrows) r1 = Mrows;
    float s = 0.f, s2 = 0.f;
    for (int r = r0; r < r1; ++r) {
        float v = pre[(size_t)r * COUT_ + c];
        s += v; s2 += v * v;
    }
    atomicAdd(&stats[c], s);
    atomicAdd(&stats[COUT_ + c], s2);
}

// BN+ReLU on stage-2 pre-activation, emit f16 NCHW for the upconv gather.
__global__ void bn_relu_y(const float* __restrict__ pre,
                          const float* __restrict__ stats,
                          const float* __restrict__ g,
                          const float* __restrict__ bt,
                          _Float16* __restrict__ yf16) {
    int idx = blockIdx.x * blockDim.x + threadIdx.x;     // P_ * 128
    if (idx >= P_ * COUT_) return;
    int co = idx & (COUT_ - 1);
    int p  = idx >> 7;
    const float invN = 1.0f / (float)P_;
    float mean = stats[co] * invN;
    float var  = stats[COUT_ + co] * invN - mean * mean;
    float sc   = g[co] * rsqrtf(var + EPS_);
    float v    = (pre[idx] - mean) * sc + bt[co];
    v = v > 0.f ? v : 0.f;
    int b = p >> 10, rem = p & 1023;
    yf16[(((size_t)b * COUT_ + co) << 10) + rem] = (_Float16)v;
}

// ===========================================================================
// Transposed-conv column gather: col[pc][r][ci*4 + jj*2 + ll]
//   r = (b*32+oy2)*32+ox2 ; iy = oy2+py-jj ; ix = ox2+px-ll
// ===========================================================================
__global__ void upcol_gather(const _Float16* __restrict__ yf16,
                             _Float16* __restrict__ col) {
    int idx = blockIdx.x * blockDim.x + threadIdx.x;     // 4 * 16384 * 128
    if (idx >= 4 * P_ * COUT_) return;
    int ci = idx & (COUT_ - 1);
    int r  = (idx >> 7) & (P_ - 1);
    int pc = idx >> 21;
    int b = r >> 10, rem = r & 1023, oy2 = rem >> 5, ox2 = rem & 31;
    int py = pc >> 1, px = pc & 1;
    const _Float16* yb = yf16 + (((size_t)b * COUT_ + ci) << 10);
    _Float16* cp = col + (((size_t)pc * P_ + r) * KUP_) + (ci << 2);
#pragma unroll
    for (int jj = 0; jj < 2; ++jj) {
#pragma unroll
        for (int ll = 0; ll < 2; ++ll) {
            int iy = oy2 + py - jj;
            int ix = ox2 + px - ll;
            _Float16 v = (_Float16)0.0f;
            if (iy >= 0 && iy < H_ && ix >= 0 && ix < W_)
                v = yb[(iy << 5) + ix];
            cp[jj * 2 + ll] = v;
        }
    }
}

// Final BN+ReLU, scatter parity-class rows back to NCHW (16,128,64,64) fp32.
__global__ void bn_relu_out(const float* __restrict__ pre,
                            const float* __restrict__ stats,
                            const float* __restrict__ g,
                            const float* __restrict__ bt,
                            float* __restrict__ out) {
    int idx = blockIdx.x * blockDim.x + threadIdx.x;     // 4*16384*128
    if (idx >= 4 * P_ * COUT_) return;
    int co = idx & (COUT_ - 1);
    int t  = idx >> 7;
    int r  = t & (P_ - 1);
    int pc = t >> 14;
    const float invN = 1.0f / (float)POUT_;
    float mean = stats[co] * invN;
    float var  = stats[COUT_ + co] * invN - mean * mean;
    float sc   = g[co] * rsqrtf(var + EPS_);
    float v    = (pre[idx] - mean) * sc + bt[co];
    v = v > 0.f ? v : 0.f;
    int b = r >> 10, rem = r & 1023, oy2 = rem >> 5, ox2 = rem & 31;
    int oy = (oy2 << 1) + (pc >> 1);
    int ox = (ox2 << 1) + (pc & 1);
    out[(((size_t)b * COUT_ + co) * 64 + oy) * 64 + ox] = v;
}

// ===========================================================================
extern "C" void kernel_launch(void* const* d_in, const int* in_sizes, int n_in,
                              void* d_out, int out_size, void* d_ws, size_t ws_size,
                              hipStream_t stream) {
    const float* x     = (const float*)d_in[0];
    const float* w_off = (const float*)d_in[1];
    const float* b_off = (const float*)d_in[2];
    const float* w_dcn = (const float*)d_in[3];
    const float* b_dcn = (const float*)d_in[4];
    const float* g1    = (const float*)d_in[5];
    const float* bt1   = (const float*)d_in[6];
    const float* w_up  = (const float*)d_in[7];
    const float* g2    = (const float*)d_in[8];
    const float* bt2   = (const float*)d_in[9];
    float* out = (float*)d_out;

    char* ws = (char*)d_ws;
    _Float16* bufA   = (_Float16*)(ws + OFF_BUFA);   // patches / samples / cols
    float*    offC   = (float*)   (ws + OFF_OFFC);
    _Float16* wOff16 = (_Float16*)(ws + OFF_WOFF);
    _Float16* wDcn16 = (_Float16*)(ws + OFF_WDCN);
    _Float16* wUp16  = (_Float16*)(ws + OFF_WUP);
    float*    yPre   = (float*)   (ws + OFF_YPRE);
    _Float16* yF16   = (_Float16*)(ws + OFF_YF16);
    float*    outPre = (float*)   (ws + OFF_OUTPRE);
    float*    stats  = (float*)   (ws + OFF_STATS);  // [0..255] stage1, [256..511] stage2

    const int TB = 256;
    auto blocks = [](long long n, int tb) { return (unsigned)((n + tb - 1) / tb); };

    // ---- prep (weights -> f16 Bt layout, zero BN stats) ----
    prep_woff<<<blocks((long long)NOFFP * KDCN_, TB), TB, 0, stream>>>(w_off, wOff16);
    prep_wdcn<<<blocks((long long)COUT_ * KDCN_, TB), TB, 0, stream>>>(w_dcn, wDcn16);
    prep_wup <<<blocks(4LL * COUT_ * KUP_, TB), TB, 0, stream>>>(w_up, wUp16);
    zero_stats<<<2, 256, 0, stream>>>(stats);

    // ---- stage 1: offset conv (im2col + WMMA GEMM, N padded to 32) ----
    im2col_x<<<blocks((long long)P_ * KDCN_, TB), TB, 0, stream>>>(x, bufA);
    gemm_wmma_f16<2><<<(P_ / 16) / 8, TB, 0, stream>>>(
        bufA, wOff16, offC, b_off, NOFF_, P_, KDCN_);

    // ---- stage 2: deformable sampling + WMMA GEMM + BN/ReLU ----
    dcn_sample<<<blocks((long long)P_ * KPIX_, TB), TB, 0, stream>>>(x, offC, bufA);
    gemm_wmma_f16<8><<<(P_ / 16) / 8, TB, 0, stream>>>(
        bufA, wDcn16, yPre, b_dcn, COUT_, P_, KDCN_);
    bn_stats<<<P_ / 64, COUT_, 0, stream>>>(yPre, P_, stats);
    bn_relu_y<<<blocks((long long)P_ * COUT_, TB), TB, 0, stream>>>(
        yPre, stats, g1, bt1, yF16);

    // ---- stage 3: transposed conv as 4 parity-class WMMA GEMMs ----
    upcol_gather<<<blocks(4LL * P_ * COUT_, TB), TB, 0, stream>>>(yF16, bufA);
    for (int pc = 0; pc < 4; ++pc) {
        gemm_wmma_f16<8><<<(P_ / 16) / 8, TB, 0, stream>>>(
            bufA + (size_t)pc * P_ * KUP_,
            wUp16 + (size_t)pc * COUT_ * KUP_,
            outPre + (size_t)pc * P_ * COUT_,
            nullptr, 0, P_, KUP_);
    }
    bn_stats<<<POUT_ / 64, COUT_, 0, stream>>>(outPre, POUT_, stats + 256);
    bn_relu_out<<<blocks(4LL * P_ * COUT_, TB), TB, 0, stream>>>(
        outPre, stats + 256, g2, bt2, out);
}
